// MultiHeadAttention_85590108275317
// MI455X (gfx1250) — matrix operand order repaired and checked by reading, not compiled, not run on hardware
//
#include <hip/hip_runtime.h>
#include <hip/hip_bf16.h>

// ---------- sizes ----------
#define BB 4
#define HH 8
#define DD 512
#define SS 2048
#define FF 4096  // HH*DD

// ---------- feature detection ----------
#if defined(__has_builtin)
#if __has_builtin(__builtin_amdgcn_global_load_async_to_lds_b128)
#define ASYNC_LDS 1
#endif
#endif
#ifndef ASYNC_LDS
#define ASYNC_LDS 0
#endif

typedef __attribute__((ext_vector_type(4))) int v4i;
#define GPTR(p) ((__attribute__((address_space(1))) v4i*)(p))
#define LPTR(p) ((__attribute__((address_space(3))) v4i*)(p))

template <int N>
__device__ __forceinline__ void wait_asynccnt() {
#if ASYNC_LDS
#if __has_builtin(__builtin_amdgcn_s_wait_asynccnt)
    __builtin_amdgcn_s_wait_asynccnt(N);
#else
    asm volatile("s_wait_asynccnt %0" :: "n"(N) : "memory");
#endif
#endif
}

// ---------- WMMA types ----------
typedef __attribute__((ext_vector_type(8)))  float   v8f;
typedef __attribute__((ext_vector_type(16))) __bf16  v16bf;

union FragU {
    v16bf bf;        // 32 bytes
    uint4 q[2];
    __device__ FragU() {}
};

__device__ __forceinline__ unsigned short f32_to_bf16(float f) {
    unsigned int u = __float_as_uint(f);
    unsigned int r = u + 0x7FFFu + ((u >> 16) & 1u);   // round-to-nearest-even
    return (unsigned short)(r >> 16);
}

__device__ __forceinline__ v8f wmma_bf16(v16bf a, v16bf b, v8f c) {
    return __builtin_amdgcn_wmma_f32_16x16x32_bf16(
        /*neg_a=*/false, a, /*neg_b=*/false, b,
        /*c_mod=*/(short)0, c, /*reuse_a=*/false, /*reuse_b=*/false);
}

// A fragment: 16(M) x 32(K) bf16, src row-major with leading dim ld (elements).
__device__ __forceinline__ v16bf load_frag_a(const unsigned short* p, int ld,
                                             int m0, int k0, int lane) {
    int r = lane & 15, h = lane >> 4;
    const unsigned short* rowp = p + (size_t)(m0 + r) * ld + k0 + h * 8;
    FragU f;
    f.q[0] = *(const uint4*)(rowp);
    f.q[1] = *(const uint4*)(rowp + 16);
    return f.bf;
}

// B fragment: 32(K) x 16(N) bf16, provided as Bt row-major N x K (ld elements).
__device__ __forceinline__ v16bf load_frag_b(const unsigned short* p, int ld,
                                             int n0, int k0, int lane) {
    int r = lane & 15, h = lane >> 4;
    const unsigned short* rowp = p + (size_t)(n0 + r) * ld + k0 + h * 8;
    FragU f;
    f.q[0] = *(const uint4*)(rowp);
    f.q[1] = *(const uint4*)(rowp + 16);
    return f.bf;
}

// ---------- conversion kernels ----------
__global__ void k_conv_xT(const float* __restrict__ x, unsigned short* __restrict__ xT) {
    // x: (B, D, S) f32 -> xT: (B, S, D) bf16
    size_t i = (size_t)blockIdx.x * 256 + threadIdx.x;
    int e = (int)(i % DD);
    size_t t = i / DD;
    int s = (int)(t % SS);
    int b = (int)(t / SS);
    xT[i] = f32_to_bf16(x[(size_t)b * DD * SS + (size_t)e * SS + s]);
}

__global__ void k_conv(const float* __restrict__ in, unsigned short* __restrict__ out) {
    size_t i = (size_t)blockIdx.x * 256 + threadIdx.x;
    out[i] = f32_to_bf16(in[i]);
}

__global__ void k_conv_WT(const float* __restrict__ W, unsigned short* __restrict__ WT) {
    // W: (F, D) f32 -> WT: (D, F) bf16
    size_t i = (size_t)blockIdx.x * 256 + threadIdx.x;
    int f = (int)(i % FF);
    int d = (int)(i / FF);
    WT[i] = f32_to_bf16(W[(size_t)f * DD + d]);
}

// ---------- generic bf16 GEMM:  C[M][N] = sum_k A[m][k] * Bt[n][k] ----------
__device__ __forceinline__ void store_out(float v, float* p)          { *p = v; }
__device__ __forceinline__ void store_out(float v, unsigned short* p) { *p = f32_to_bf16(v); }

template <typename OUTT>
__global__ __launch_bounds__(256)
void gemm_bt(const unsigned short* __restrict__ A,
             const unsigned short* __restrict__ Bt,
             OUTT* __restrict__ C,
             int M, int N, int K,
             long long offAb, long long offAh,
             long long offBb, long long offBh,
             long long offCb, long long offCh,
             int Hdiv) {
    int z  = blockIdx.z;
    int zb = z / Hdiv, zh = z % Hdiv;
    A  += (size_t)zb * offAb + (size_t)zh * offAh;
    Bt += (size_t)zb * offBb + (size_t)zh * offBh;
    C  += (size_t)zb * offCb + (size_t)zh * offCh;

    __shared__ unsigned short As[2][128 * 32];
    __shared__ unsigned short Bs[2][128 * 32];

    int tid = threadIdx.x;
    int lane = tid & 31, wave = tid >> 5;      // 8 waves (wave32)
    int wm = wave >> 1, wn = wave & 1;         // 4x2 wave grid: 32x64 wave tile
    int m0 = blockIdx.y * 128, n0 = blockIdx.x * 128;

    int lrow = tid >> 1;                       // 0..127
    int lcol = (tid & 1) * 16;                 // 0 or 16 (each thread moves 16 elems = 32B)

    v8f acc[2][4];
    for (int i = 0; i < 2; ++i)
        for (int j = 0; j < 4; ++j) acc[i][j] = v8f{};

    // stage one 128x32 A-tile + B-tile into LDS buffer `buf` (4 x b128 / thread)
    auto stage = [&](int buf, int k0) {
        const unsigned short* gA = A  + (size_t)(m0 + lrow) * K + k0 + lcol;
        const unsigned short* gB = Bt + (size_t)(n0 + lrow) * K + k0 + lcol;
        unsigned short* sA = &As[buf][lrow * 32 + lcol];
        unsigned short* sB = &Bs[buf][lrow * 32 + lcol];
#if ASYNC_LDS
        __builtin_amdgcn_global_load_async_to_lds_b128(GPTR(gA),     LPTR(sA),     0, 0);
        __builtin_amdgcn_global_load_async_to_lds_b128(GPTR(gA + 8), LPTR(sA + 8), 0, 0);
        __builtin_amdgcn_global_load_async_to_lds_b128(GPTR(gB),     LPTR(sB),     0, 0);
        __builtin_amdgcn_global_load_async_to_lds_b128(GPTR(gB + 8), LPTR(sB + 8), 0, 0);
#else
        *(uint4*)(sA)     = *(const uint4*)(gA);
        *(uint4*)(sA + 8) = *(const uint4*)(gA + 8);
        *(uint4*)(sB)     = *(const uint4*)(gB);
        *(uint4*)(sB + 8) = *(const uint4*)(gB + 8);
#endif
    };

    const int KT = K >> 5;
    stage(0, 0);
    for (int kt = 0; kt < KT; ++kt) {
        const int cur = kt & 1;
        const bool more = (kt + 1 < KT);
        if (more) {
            stage(1 - cur, (kt + 1) * 32);     // overlap next tile with compute
            wait_asynccnt<4>();                // current tile's 4 loads complete
        } else {
            wait_asynccnt<0>();
        }
        __syncthreads();
        for (int i = 0; i < 2; ++i) {
            v16bf a = load_frag_a(&As[cur][0], 32, wm * 32 + i * 16, 0, lane);
            for (int j = 0; j < 4; ++j) {
                v16bf b = load_frag_b(&Bs[cur][0], 32, wn * 64 + j * 16, 0, lane);
                acc[i][j] = wmma_bf16(a, b, acc[i][j]);
            }
        }
        __syncthreads();
    }

    int r = lane & 15, hh = lane >> 4;
    for (int i = 0; i < 2; ++i)
        for (int j = 0; j < 4; ++j) {
            int row = m0 + wm * 32 + i * 16 + hh * 8;
            int col = n0 + wn * 64 + j * 16 + r;
            for (int e = 0; e < 8; ++e)
                store_out(acc[i][j][e], &C[(size_t)(row + e) * N + col]);
        }
}

// ---------- flash attention ----------
// scores St[t][s] = (1/sqrt(D)) * sum_e Kt[t][e]*Qt[s][e]; softmax over s;
// Ot[t][e] = sum_s P[t][s]*V[e][s];  written to Ocat[b][t][h*D+e] (bf16).
__global__ __launch_bounds__(128)
void k_attn(const unsigned short* __restrict__ Qt,
            const unsigned short* __restrict__ Kt,
            const unsigned short* __restrict__ V,
            unsigned short* __restrict__ Ocat) {
    const int t0 = blockIdx.x * 16;
    const int h  = blockIdx.y;
    const int b  = blockIdx.z;
    const size_t bh = (size_t)b * HH + h;
    const unsigned short* Qbh = Qt + bh * ((size_t)SS * DD);
    const unsigned short* Kbh = Kt + bh * ((size_t)SS * DD);
    const unsigned short* Vbh = V  + bh * ((size_t)DD * SS);
    unsigned short* Ob = Ocat + (size_t)b * SS * FF;

    __shared__ unsigned short Pl[16 * 128];     // P tile, A-layout for O-GEMM
    __shared__ float redm[4][16];
    __shared__ float reds[4][16];

    const int tid = threadIdx.x;
    const int lane = tid & 31, wave = tid >> 5; // 4 waves
    const int r = lane & 15, hhalf = lane >> 4;

    v8f O[8];
    for (int j = 0; j < 8; ++j) O[j] = v8f{};
    float m8[8], l8[8];
    for (int j = 0; j < 8; ++j) { m8[j] = -1e30f; l8[j] = 0.f; }

    const float scale = 0.044194173824159216f;  // 1/sqrt(512)
    const float L2E   = 1.4426950408889634f;

    for (int s0 = 0; s0 < SS; s0 += 128) {
        // --- scores: this wave's 32-wide s slice ---
        v8f Sa[2];
        Sa[0] = v8f{}; Sa[1] = v8f{};
        const int ssub = s0 + wave * 32;
        for (int e0 = 0; e0 < DD; e0 += 32) {
            v16bf a  = load_frag_a(Kbh, DD, t0, e0, lane);
            v16bf b0 = load_frag_b(Qbh, DD, ssub, e0, lane);
            v16bf b1 = load_frag_b(Qbh, DD, ssub + 16, e0, lane);
            Sa[0] = wmma_bf16(a, b0, Sa[0]);
            Sa[1] = wmma_bf16(a, b1, Sa[1]);
        }
        // --- scale + partial row max (rows live per VGPR; cols across lanes) ---
        float pm[8];
        for (int j = 0; j < 8; ++j) {
            Sa[0][j] *= scale; Sa[1][j] *= scale;
            pm[j] = fmaxf(Sa[0][j], Sa[1][j]);
        }
        for (int off = 1; off < 16; off <<= 1)
            for (int j = 0; j < 8; ++j)
                pm[j] = fmaxf(pm[j], __shfl_xor(pm[j], off, 32));
        if (r == 0)
            for (int j = 0; j < 8; ++j) redm[wave][hhalf * 8 + j] = pm[j];
        __syncthreads();
        float mt[8], al[8];
        for (int j = 0; j < 8; ++j) {
            float bm = fmaxf(fmaxf(redm[0][hhalf * 8 + j], redm[1][hhalf * 8 + j]),
                             fmaxf(redm[2][hhalf * 8 + j], redm[3][hhalf * 8 + j]));
            mt[j] = fmaxf(m8[j], bm);
            al[j] = exp2f((m8[j] - mt[j]) * L2E);
        }
        __syncthreads();
        // --- P = exp(S - mt) -> LDS (layout transpose) + partial row sums ---
        float ps[8];
        for (int j = 0; j < 8; ++j) ps[j] = 0.f;
        for (int jt = 0; jt < 2; ++jt)
            for (int j = 0; j < 8; ++j) {
                float p = exp2f((Sa[jt][j] - mt[j]) * L2E);
                ps[j] += p;
                Pl[(hhalf * 8 + j) * 128 + wave * 32 + jt * 16 + r] = f32_to_bf16(p);
            }
        for (int off = 1; off < 16; off <<= 1)
            for (int j = 0; j < 8; ++j) ps[j] += __shfl_xor(ps[j], off, 32);
        if (r == 0)
            for (int j = 0; j < 8; ++j) reds[wave][hhalf * 8 + j] = ps[j];
        __syncthreads();
        for (int j = 0; j < 8; ++j) {
            float bs = reds[0][hhalf * 8 + j] + reds[1][hhalf * 8 + j] +
                       reds[2][hhalf * 8 + j] + reds[3][hhalf * 8 + j];
            l8[j] = l8[j] * al[j] + bs;
            m8[j] = mt[j];
        }
        // --- rescale running O, then O += P * V (this wave owns 128 e-cols) ---
        for (int jn = 0; jn < 8; ++jn)
            for (int j = 0; j < 8; ++j) O[jn][j] *= al[j];
        const int e0w = wave * 128;
        for (int ks = 0; ks < 4; ++ks) {
            v16bf a = load_frag_a(Pl, 128, 0, ks * 32, lane);
            for (int jn = 0; jn < 8; ++jn) {
                v16bf bv = load_frag_b(Vbh, SS, e0w + jn * 16, s0 + ks * 32, lane);
                O[jn] = wmma_bf16(a, bv, O[jn]);
            }
        }
        __syncthreads();   // Pl / reductions reused next s-block
    }

    // --- epilogue: O /= l, write Ocat[b][t][h*D + e] ---
    for (int jn = 0; jn < 8; ++jn)
        for (int j = 0; j < 8; ++j) {
            float v = O[jn][j] / l8[j];
            int row = t0 + hhalf * 8 + j;
            int col = h * DD + wave * 128 + jn * 16 + r;
            Ob[(size_t)row * FF + col] = f32_to_bf16(v);
        }
}

// ---------- host launcher ----------
extern "C" void kernel_launch(void* const* d_in, const int* in_sizes, int n_in,
                              void* d_out, int out_size, void* d_ws, size_t ws_size,
                              hipStream_t stream) {
    const float* x  = (const float*)d_in[0];
    const float* Wq = (const float*)d_in[1];
    const float* Wk = (const float*)d_in[2];
    const float* Wv = (const float*)d_in[3];
    const float* Wo = (const float*)d_in[4];
    float* out = (float*)d_out;

    char* ws = (char*)d_ws;
    size_t off = 0;
    auto alloc_us = [&](size_t elems) -> unsigned short* {
        unsigned short* p = (unsigned short*)(ws + off);
        off += (elems * 2 + 255) & ~(size_t)255;
        return p;
    };

    unsigned short* xT   = alloc_us((size_t)BB * SS * DD);
    unsigned short* Wqb  = alloc_us((size_t)HH * DD * DD);
    unsigned short* Wkb  = alloc_us((size_t)HH * DD * DD);
    unsigned short* Wvb  = alloc_us((size_t)HH * DD * DD);
    unsigned short* WTb  = alloc_us((size_t)DD * FF);
    unsigned short* Qtb  = alloc_us((size_t)BB * HH * SS * DD);
    unsigned short* Ktb  = alloc_us((size_t)BB * HH * SS * DD);
    unsigned short* Vb   = alloc_us((size_t)BB * HH * DD * SS);
    unsigned short* Ocat = alloc_us((size_t)BB * SS * FF);

    // 1) conversions / transposes
    k_conv_xT<<<(BB * SS * DD) / 256, 256, 0, stream>>>(x, xT);
    k_conv<<<(HH * DD * DD) / 256, 256, 0, stream>>>(Wq, Wqb);
    k_conv<<<(HH * DD * DD) / 256, 256, 0, stream>>>(Wk, Wkb);
    k_conv<<<(HH * DD * DD) / 256, 256, 0, stream>>>(Wv, Wvb);
    k_conv_WT<<<(DD * FF) / 256, 256, 0, stream>>>(Wo, WTb);

    const long long SD = (long long)SS * DD;      // 2048*512
    const long long D2 = (long long)DD * DD;      // 512*512

    // 2) Qt[b,h] (SxD) = xT[b] (SxD) x Wq[h]^T   (Bt = Wq row-major)
    gemm_bt<unsigned short><<<dim3(DD / 128, SS / 128, BB * HH), 256, 0, stream>>>(
        xT, Wqb, Qtb, SS, DD, DD,
        SD, 0, 0, D2, (long long)HH * SD, SD, HH);
    // 3) Kt
    gemm_bt<unsigned short><<<dim3(DD / 128, SS / 128, BB * HH), 256, 0, stream>>>(
        xT, Wkb, Ktb, SS, DD, DD,
        SD, 0, 0, D2, (long long)HH * SD, SD, HH);
    // 4) V[b,h] (DxS) = Wv[h] (DxD) x x[b]  (Bt = xT)
    gemm_bt<unsigned short><<<dim3(SS / 128, DD / 128, BB * HH), 256, 0, stream>>>(
        Wvb, xT, Vb, DD, SS, DD,
        0, D2, SD, 0, (long long)HH * SD, SD, HH);

    // 5) flash attention -> Ocat[b] (S x F) bf16
    k_attn<<<dim3(SS / 16, HH, BB), 128, 0, stream>>>(Qtb, Ktb, Vb, Ocat);

    // 6) y[b] (DxS) f32 = WT (DxF) x Ocat[b]^T-as-Bt
    gemm_bt<float><<<dim3(SS / 128, DD / 128, BB), 256, 0, stream>>>(
        WTb, Ocat, out, DD, SS, FF,
        0, 0, (long long)SS * FF, 0, (long long)DD * SS, 0, 1);
}
